// LSTM_71390946394820
// MI455X (gfx1250) — matrix operand order repaired and checked by reading, not compiled, not run on hardware
//
#include <hip/hip_runtime.h>
#include <hip/hip_bf16.h>

// ---------------------------------------------------------------------------
// LSTM on MI455X (gfx1250): persistent wave32 WMMA kernel.
//   B=128, T=512, D=256, H=512, C=128.
//   z_t = [x_t | h_t] @ [Wx;Wh] + b   (K = 768, N = 4H = 2048, M = B = 128)
//   f16 operands, f32 accumulation via v_wmma_f32_16x16x32_f16.
//   Time-invariant B fragments staged once in LDS (96 KB / workgroup).
// ---------------------------------------------------------------------------

#define BATCH 128
#define TSTEPS 512
#define DIN 256
#define HID 512
#define COUT 128
#define KDIM (DIN + HID)          // 768
#define KSTEPS (KDIM / 32)        // 24
#define NWG 32                    // persistent workgroups
#define WGTHREADS 256             // 8 waves (wave32)
#define ASTRIDE (BATCH * KDIM)    // halves per A buffer
#define LDSB_BYTES (4 * KSTEPS * 32 * 32)   // 4 ct * 24 ks * 1 KiB frag = 96 KiB

typedef _Float16 hv16 __attribute__((ext_vector_type(16)));
typedef _Float16 hv8  __attribute__((ext_vector_type(8)));
typedef _Float16 hv4  __attribute__((ext_vector_type(4)));
typedef float    fv8  __attribute__((ext_vector_type(8)));

union F16x16 { hv16 v; hv8 h[2]; };

__device__ __forceinline__ float sigm(float x) {
    return 1.0f / (1.0f + __expf(-x));
}
__device__ __forceinline__ float tanh_fast(float x) {
    return 2.0f / (1.0f + __expf(-2.0f * x)) - 1.0f;
}

// ---------------------------------------------------------------------------
// Prep 1: W^T in f16: wcatT[n][k], n in [0,2048) over gates [g,i,f,o],
// k in [0,768) = [Wx rows | Wh rows]; row-contiguous in k.
// ---------------------------------------------------------------------------
__global__ void build_wcatT(const float* __restrict__ gx, const float* __restrict__ gh,
                            const float* __restrict__ ix, const float* __restrict__ ih,
                            const float* __restrict__ fx, const float* __restrict__ fh,
                            const float* __restrict__ ox, const float* __restrict__ oh,
                            _Float16* __restrict__ wcatT) {
    int idx = blockIdx.x * blockDim.x + threadIdx.x;   // over 2048*768
    int n = idx / KDIM;
    int k = idx - n * KDIM;
    int gate = n >> 9;          // /512
    int col  = n & 511;
    const float* wx = (gate == 0) ? gx : (gate == 1) ? ix : (gate == 2) ? fx : ox;
    const float* wh = (gate == 0) ? gh : (gate == 1) ? ih : (gate == 2) ? fh : oh;
    float v = (k < DIN) ? wx[k * HID + col] : wh[(k - DIN) * HID + col];
    wcatT[idx] = (_Float16)v;
}

// ---------------------------------------------------------------------------
// Prep 2: A[0] = [f16(x_0) | h_0 = 0], row-major [128][768].
// ---------------------------------------------------------------------------
__global__ void init_A0(const float* __restrict__ x, _Float16* __restrict__ A0) {
    int idx = blockIdx.x * blockDim.x + threadIdx.x;   // over 128*768
    int m = idx / KDIM;
    int k = idx - m * KDIM;
    float v = (k < DIN) ? x[(size_t)m * (TSTEPS * DIN) + k] : 0.0f;
    A0[idx] = (_Float16)v;
}

// ---------------------------------------------------------------------------
// Persistent LSTM kernel. 32 workgroups x 256 threads (8 waves).
//   wg owns hidden cols [wg*16, wg*16+16); wave w owns rows [16w, 16w+16).
//   acc tile ct (0..3) == gate ct.  c state stays in VGPRs for all 512 steps.
//   B fragments (time-invariant) live in LDS; A streams from L2.
// ---------------------------------------------------------------------------
__global__ void __launch_bounds__(WGTHREADS)
lstm_persistent(const float* __restrict__ x,
                const _Float16* __restrict__ wcatT,
                const float* __restrict__ gb, const float* __restrict__ ib,
                const float* __restrict__ fb, const float* __restrict__ ob,
                _Float16* __restrict__ Abase,
                unsigned* __restrict__ counter) {
    extern __shared__ _Float16 ldsB[];     // [ct][ks][chunk][lane][8 halves]

    const int wg   = blockIdx.x;       // 0..31
    const int tid  = threadIdx.x;
    const int w    = tid >> 5;         // wave id 0..7
    const int l    = tid & 31;         // lane (wave32)
    const int lrow = l & 15;
    const int lhi  = l >> 4;
    const int m0   = w * 16;

    // Per-lane gate biases: bias depends only on N = wg*16 + (lane%16).
    const int ncol = wg * 16 + lrow;
    const float b0 = gb[ncol], b1 = ib[ncol], b2 = fb[ncol], b3 = ob[ncol];

    // ---- Stage time-invariant B fragments into LDS (once) ----
    // Fragment f = ct*KSTEPS + ks, 1 KiB each: two 512-B chunks, lane-major
    // 16-B stride (conflict-free ds_load_b128 / ds_store_b128).
    for (int i = 0; i < (4 * KSTEPS) / 8; ++i) {       // 12 frags per wave
        int f  = w + 8 * i;
        int ct = f / KSTEPS;
        int ks = f - ct * KSTEPS;
        const _Float16* bp = wcatT + (size_t)(ct * HID + ncol) * KDIM + 8 * lhi + ks * 32;
        hv8 lo = *(const hv8*)(bp);
        hv8 hi = *(const hv8*)(bp + 16);
        _Float16* dst = ldsB + (size_t)f * 512 + l * 8;
        *(hv8*)(dst)       = lo;
        *(hv8*)(dst + 256) = hi;
    }
    __syncthreads();

    // Per-ct LDS base pointers keep DS immediate offsets < 24 KiB.
    const _Float16* lB0 = ldsB + 0 * (KSTEPS * 512) + l * 8;
    const _Float16* lB1 = ldsB + 1 * (KSTEPS * 512) + l * 8;
    const _Float16* lB2 = ldsB + 2 * (KSTEPS * 512) + l * 8;
    const _Float16* lB3 = ldsB + 3 * (KSTEPS * 512) + l * 8;

    fv8 c = {};                        // cell state, resident in VGPRs

    for (int t = 0; t < TSTEPS; ++t) {
        // Double-buffered A: arithmetic off one kernarg => global_load path.
        const _Float16* A  = Abase + (size_t)(t & 1) * ASTRIDE;
        _Float16*       An = Abase + (size_t)((t + 1) & 1) * ASTRIDE;

        fv8 acc0 = {}, acc1 = {}, acc2 = {}, acc3 = {};
        // A fragment base: 16-bit A 16x32 layout -> lane covers K = 8*lhi..+7, +16.
        const _Float16* aRow = A + (size_t)(m0 + lrow) * KDIM + 8 * lhi;

#pragma unroll 4
        for (int ks = 0; ks < KSTEPS; ++ks) {
            F16x16 a;
            a.h[0] = *(const hv8*)(aRow + ks * 32);
            a.h[1] = *(const hv8*)(aRow + ks * 32 + 16);

            F16x16 bf0, bf1, bf2, bf3;
            bf0.h[0] = *(const hv8*)(lB0 + ks * 512);
            bf0.h[1] = *(const hv8*)(lB0 + ks * 512 + 256);
            bf1.h[0] = *(const hv8*)(lB1 + ks * 512);
            bf1.h[1] = *(const hv8*)(lB1 + ks * 512 + 256);
            bf2.h[0] = *(const hv8*)(lB2 + ks * 512);
            bf2.h[1] = *(const hv8*)(lB2 + ks * 512 + 256);
            bf3.h[0] = *(const hv8*)(lB3 + ks * 512);
            bf3.h[1] = *(const hv8*)(lB3 + ks * 512 + 256);

            acc0 = __builtin_amdgcn_wmma_f32_16x16x32_f16(false, a.v, false, bf0.v,
                                                          (short)0, acc0, false, false);
            acc1 = __builtin_amdgcn_wmma_f32_16x16x32_f16(false, a.v, false, bf1.v,
                                                          (short)0, acc1, false, false);
            acc2 = __builtin_amdgcn_wmma_f32_16x16x32_f16(false, a.v, false, bf2.v,
                                                          (short)0, acc2, false, false);
            acc3 = __builtin_amdgcn_wmma_f32_16x16x32_f16(false, a.v, false, bf3.v,
                                                          (short)0, acc3, false, false);
        }

        // Gate math + state update; write h_{t+1} (f16) into the next A buffer.
        // C/D layout: VGPR r, lane l -> (M = r + 8*(l>>4), N = l&15).
        _Float16* hDst = An + (size_t)(m0 + 8 * lhi) * KDIM + DIN + wg * 16 + lrow;
#pragma unroll
        for (int r = 0; r < 8; ++r) {
            float g  = tanh_fast(acc0[r] + b0);
            float ii = sigm(acc1[r] + b1);
            float ff = sigm(acc2[r] + b2);
            float oo = sigm(acc3[r] + b3);
            float cn = g * ii + c[r] * ff;
            c[r] = cn;
            float hh = tanh_fast(cn) * oo;
            hDst[(size_t)r * KDIM] = (_Float16)hh;
        }

        // Convert x_{t+1} into the next A buffer (32768 elems over 8192 threads).
        if (t < TSTEPS - 1) {
            int base = (wg * WGTHREADS + tid) * 4;       // [0, 32768)
            int m = base >> 8;                            // /256
            int d = base & 255;
            const float* xp = x + (size_t)m * (TSTEPS * DIN) + (size_t)(t + 1) * DIN + d;
            float4 v = *(const float4*)xp;
            hv4 p;
            p[0] = (_Float16)v.x; p[1] = (_Float16)v.y;
            p[2] = (_Float16)v.z; p[3] = (_Float16)v.w;
            *(hv4*)(An + (size_t)m * KDIM + d) = p;
        }

        // -------- grid barrier (release -> arrive -> spin -> acquire) --------
        __threadfence();
        __syncthreads();
        if (tid == 0) {
            atomicAdd(counter, 1u);
            const unsigned target = (unsigned)(t + 1) * NWG;
            while (__hip_atomic_load(counter, __ATOMIC_RELAXED,
                                     __HIP_MEMORY_SCOPE_AGENT) < target) {
                __builtin_amdgcn_s_sleep(2);
            }
        }
        __syncthreads();
        __threadfence();
    }
}

// ---------------------------------------------------------------------------
// Final projection: out[b][cc] = h_T[b][:] @ ph + pb.  16.7 MFLOP — trivial.
// h_T lives in A buffer 0 (step 511 wrote buffer (511+1)&1 == 0), cols [256,768).
// ---------------------------------------------------------------------------
__global__ void proj_out(const _Float16* __restrict__ A0,
                         const float* __restrict__ ph, const float* __restrict__ pb,
                         float* __restrict__ out) {
    int idx = blockIdx.x * blockDim.x + threadIdx.x;   // over 128*128
    int b  = idx >> 7;
    int cc = idx & 127;
    const _Float16* h = A0 + (size_t)b * KDIM + DIN;
    float s = pb[cc];
#pragma unroll 8
    for (int k = 0; k < HID; ++k)
        s += (float)h[k] * ph[(size_t)k * COUT + cc];
    out[idx] = s;
}

// ---------------------------------------------------------------------------
extern "C" void kernel_launch(void* const* d_in, const int* in_sizes, int n_in,
                              void* d_out, int out_size, void* d_ws, size_t ws_size,
                              hipStream_t stream) {
    const float* x  = (const float*)d_in[0];
    const float* gx = (const float*)d_in[1];
    const float* gh = (const float*)d_in[2];
    const float* gb = (const float*)d_in[3];
    const float* ix = (const float*)d_in[4];
    const float* ih = (const float*)d_in[5];
    const float* ib = (const float*)d_in[6];
    const float* fx = (const float*)d_in[7];
    const float* fh = (const float*)d_in[8];
    const float* fb = (const float*)d_in[9];
    const float* ox = (const float*)d_in[10];
    const float* oh = (const float*)d_in[11];
    const float* ob = (const float*)d_in[12];
    const float* ph = (const float*)d_in[13];
    const float* pb = (const float*)d_in[14];

    char* ws = (char*)d_ws;
    unsigned*  counter = (unsigned*)ws;                         // 256 B
    _Float16*  wcatT   = (_Float16*)(ws + 256);                 // 2048*768*2 = 3 MiB
    _Float16*  Abase   = (_Float16*)(ws + 256 + (size_t)2048 * KDIM * 2); // 2 x 192 KiB

    // Barrier counter must start at 0 every launch (capture-legal stream op).
    hipMemsetAsync(ws, 0, 256, stream);

    build_wcatT<<<(2048 * KDIM) / 256, 256, 0, stream>>>(gx, gh, ix, ih, fx, fh, ox, oh, wcatT);
    init_A0<<<(BATCH * KDIM) / 256, 256, 0, stream>>>(x, Abase);
    lstm_persistent<<<NWG, WGTHREADS, LDSB_BYTES, stream>>>(x, wcatT, gb, ib, fb, ob,
                                                            Abase, counter);
    proj_out<<<(BATCH * COUT) / 256, 256, 0, stream>>>(Abase, ph, pb, (float*)d_out);
}
